// Qwen3MoeAttention_24833500906103
// MI455X (gfx1250) — matrix operand order repaired and checked by reading, not compiled, not run on hardware
//
#include <hip/hip_runtime.h>

#define T_TOK   2048
#define HIDDEN  2048
#define NHEADS  32
#define NKV     4
#define HDIM    128
#define QSIZE   (NHEADS * HDIM)     // 4096
#define KVSIZE  (NKV * HDIM)        // 512
#define QKV_N   (QSIZE + 2 * KVSIZE)// 5120

typedef __bf16 bf16_t;
typedef __attribute__((ext_vector_type(4)))  bf16_t v4bf;
typedef __attribute__((ext_vector_type(8)))  bf16_t v8bf;
typedef __attribute__((ext_vector_type(16))) bf16_t v16bf;
typedef __attribute__((ext_vector_type(8)))  float  v8f;
typedef __attribute__((ext_vector_type(4)))  unsigned int v4u32;
typedef __attribute__((ext_vector_type(8)))  int v8i32;
typedef __attribute__((ext_vector_type(4)))  int v4i32;

#if defined(__has_builtin)
#if __has_builtin(__builtin_amdgcn_tensor_load_to_lds) && __has_builtin(__builtin_amdgcn_s_wait_tensorcnt)
#define USE_TDM 1
#endif
#endif
#ifndef USE_TDM
#define USE_TDM 0
#endif

__device__ __forceinline__ bf16_t f2bf(float f) {
    union { float f; unsigned int u; } a; a.f = f;
    unsigned int u = a.u;
    unsigned int r = u + 0x7FFFu + ((u >> 16) & 1u);   // round to nearest even
    union { unsigned short s; bf16_t b; } o; o.s = (unsigned short)(r >> 16);
    return o.b;
}

__device__ __forceinline__ v8f wmma_bf16(v16bf a, v16bf b, v8f c) {
    return __builtin_amdgcn_wmma_f32_16x16x32_bf16(false, a, false, b, (short)0, c, false, false);
}

// A-fragment (16x32 bf16): lane m holds row m; lanes 0-15: K 0..7 & 16..23; lanes 16-31: K 8..15 & 24..31
__device__ __forceinline__ v16bf frag_a(const bf16_t* row_base, int aoff /* (lane&16)?8:0 */) {
    v8bf lo = *(const v8bf*)(row_base + aoff);
    v8bf hi = *(const v8bf*)(row_base + 16 + aoff);
    return __builtin_shufflevector(lo, hi, 0,1,2,3,4,5,6,7,8,9,10,11,12,13,14,15);
}

// B-fragment (32x16 bf16): lane n holds column n; lanes 0-15: K 0..7 & 16..23; lanes 16-31: K 8..15 & 24..31
__device__ __forceinline__ v16bf frag_b(const bf16_t* col_base, int bsel /* (lane&16)?16:0 */) {
    v8bf lo = *(const v8bf*)(col_base + (bsel >> 1));
    v8bf hi = *(const v8bf*)(col_base + (bsel >> 1) + 16);
    return __builtin_shufflevector(lo, hi, 0,1,2,3,4,5,6,7,8,9,10,11,12,13,14,15);
}

#if USE_TDM
// Issue one TDM 2D tile load (bf16 elements) into LDS with hardware padding.
// tile_d0 elems per row, tile_d1 rows, row stride `stride0` elems.
// pad codes per ISA Group1: interval 0=2DW..7=256DW, amount 0=1DW..127=128DW.
__device__ __forceinline__ void tdm_load_2d(unsigned lds_addr, const void* gptr,
                                            unsigned tile_d0, unsigned tile_d1,
                                            unsigned stride0,
                                            unsigned pad_interval_code,
                                            unsigned pad_amount_code) {
    unsigned long long ga = (unsigned long long)(size_t)gptr;
    v4u32 g0;
    g0[0] = 1u;                                            // count=1, user descriptor
    g0[1] = lds_addr;                                      // LDS byte address
    g0[2] = (unsigned)(ga & 0xFFFFFFFFu);                  // global_addr[31:0]
    g0[3] = (unsigned)((ga >> 32) & 0x1FFFFFFu) | (2u << 30); // global_addr[56:32], type=2
    v8i32 g1;
    g1[0] = (int)((1u << 16)                               // data_size = 2 bytes
                | (1u << 20)                               // pad_enable
                | (pad_interval_code << 22)
                | (pad_amount_code << 25));
    g1[1] = (int)((tile_d0 & 0xFFFFu) << 16);              // tensor_dim0[15:0]
    g1[2] = (int)(((tile_d0 >> 16) & 0xFFFFu) | ((tile_d1 & 0xFFFFu) << 16)); // dim0 hi | dim1 lo
    g1[3] = (int)(((tile_d1 >> 16) & 0xFFFFu) | ((tile_d0 & 0xFFFFu) << 16)); // dim1 hi | tile_dim0
    g1[4] = (int)(tile_d1 & 0xFFFFu);                      // tile_dim1, tile_dim2 = 0
    g1[5] = (int)stride0;                                  // tensor_dim0_stride[31:0]
    g1[6] = 0;                                             // stride0 hi | stride1 lo
    g1[7] = 0;                                             // stride1 hi
    v4i32 zz4 = {0, 0, 0, 0};
    v8i32 zz8 = {0, 0, 0, 0, 0, 0, 0, 0};
    __builtin_amdgcn_tensor_load_to_lds(g0, g1, zz4, zz4, zz8, 0);
}
#endif

// ---------------------------------------------------------------- fp32 -> bf16
__global__ __launch_bounds__(256) void cvt_f32_bf16(const float* __restrict__ in,
                                                    bf16_t* __restrict__ out, int n4) {
    int i = blockIdx.x * blockDim.x + threadIdx.x;
    int stride = gridDim.x * blockDim.x;
    const float4* in4 = (const float4*)in;
    v4bf* out4 = (v4bf*)out;
    for (; i < n4; i += stride) {
        float4 v = in4[i];
        v4bf o;
        o[0] = f2bf(v.x); o[1] = f2bf(v.y); o[2] = f2bf(v.z); o[3] = f2bf(v.w);
        out4[i] = o;
    }
}

// ------------------------------------------------- C[M,N] = A[M,K] * B[N,K]^T
// 128x128 block tile, 8 waves (2M x 4N), wave tile 64x32.
// K-step 64, double-buffered LDS, tiles staged by the Tensor Data Mover
// (TDM pads each 64-elem row by 8 halves -> LDS stride 72).
__global__ __launch_bounds__(256) void gemm_bf16_nt(const bf16_t* __restrict__ A,
                                                    const bf16_t* __restrict__ B,
                                                    float* __restrict__ C,
                                                    int M, int N, int K) {
    __shared__ bf16_t As[2][128 * 72]; // [m][k], pad stride 72 halves
    __shared__ bf16_t Bs[2][128 * 72]; // [n][k], pad stride 72 halves
    const int tid = threadIdx.x, lane = tid & 31, wave = tid >> 5;
    const int m0 = blockIdx.y * 128, n0 = blockIdx.x * 128;
    const int wm = (wave >> 2) * 64, wn = (wave & 3) * 32;
    const int aoff = (lane & 16) ? 8 : 0;
    const int bsel = (lane & 16) ? 16 : 0;
    const int mrow = lane & 15;

    v8f z = {0.f,0.f,0.f,0.f,0.f,0.f,0.f,0.f};
    v8f acc[4][2];
    for (int mi = 0; mi < 4; ++mi) for (int ni = 0; ni < 2; ++ni) acc[mi][ni] = z;

    auto stage = [&](int buf, int k0) {
#if USE_TDM
        if (tid < 32) {   // one wave issues both tile DMAs (wave-level ops)
            tdm_load_2d((unsigned)(size_t)&As[buf][0], &A[(size_t)m0 * K + k0],
                        64u, 128u, (unsigned)K, 4u /*32 DW*/, 3u /*4 DW*/);
            tdm_load_2d((unsigned)(size_t)&Bs[buf][0], &B[(size_t)n0 * K + k0],
                        64u, 128u, (unsigned)K, 4u, 3u);
        }
        if (k0 + 64 < K) { // L2 prefetch of the tile after this one
            int row = tid & 127;
            const bf16_t* g = (tid < 128) ? &A[(size_t)(m0 + row) * K + k0 + 64]
                                          : &B[(size_t)(n0 + row) * K + k0 + 64];
            __builtin_prefetch(g, 0, 1);
        }
#else
        for (int i = 0; i < 4; ++i) {
            int idx = tid + i * 256;           // 1024 chunks of 8 halves
            int row = idx >> 3, c = idx & 7;
            *(v8bf*)&As[buf][row * 72 + c * 8] = *(const v8bf*)&A[(size_t)(m0 + row) * K + k0 + c * 8];
            *(v8bf*)&Bs[buf][row * 72 + c * 8] = *(const v8bf*)&B[(size_t)(n0 + row) * K + k0 + c * 8];
        }
#endif
    };

    const int nk = K >> 6;
    stage(0, 0);
    for (int kt = 0; kt < nk; ++kt) {
#if USE_TDM
        __builtin_amdgcn_s_wait_tensorcnt(0);  // issuing wave drains its TDM ops
#endif
        __syncthreads();
        if (kt + 1 < nk) stage((kt + 1) & 1, (kt + 1) * 64);
        const bf16_t* Ab = As[kt & 1];
        const bf16_t* Bb = Bs[kt & 1];
        for (int kc = 0; kc < 2; ++kc) {
            const int off = kc * 32;
            v16bf af[4], bfr[2];
            for (int mi = 0; mi < 4; ++mi)
                af[mi] = frag_a(&Ab[(wm + mi * 16 + mrow) * 72 + off], aoff);
            for (int ni = 0; ni < 2; ++ni)
                bfr[ni] = frag_b(&Bb[(wn + ni * 16 + mrow) * 72 + off], bsel);
            for (int mi = 0; mi < 4; ++mi)
                for (int ni = 0; ni < 2; ++ni)
                    acc[mi][ni] = wmma_bf16(af[mi], bfr[ni], acc[mi][ni]);
        }
    }

    const int rowadd = (lane & 16) ? 8 : 0;
    for (int mi = 0; mi < 4; ++mi)
        for (int ni = 0; ni < 2; ++ni) {
            union { v8f v; float f[8]; } u; u.v = acc[mi][ni];
            int col = n0 + wn + ni * 16 + mrow;
            int rbase = m0 + wm + mi * 16 + rowadd;
            for (int j = 0; j < 8; ++j)
                C[(size_t)(rbase + j) * N + col] = u.f[j];
        }
}

// ---------------------------------------- per-head RMSNorm + RoPE, fp32->bf16
__device__ __forceinline__ void rmsrope_store(const float* __restrict__ x,
                                              const float* __restrict__ w,
                                              bf16_t* __restrict__ out, int lane,
                                              float c0, float s0, float c1, float s1) {
    float x0 = x[lane], x1 = x[lane + 32], x2 = x[lane + 64], x3 = x[lane + 96];
    float ss = x0 * x0 + x1 * x1 + x2 * x2 + x3 * x3;
    for (int m = 1; m < 32; m <<= 1) ss += __shfl_xor(ss, m, 32);
    float rstd = rsqrtf(ss * (1.0f / 128.0f) + 1e-6f);
    float n0 = x0 * rstd * w[lane],      n1 = x1 * rstd * w[lane + 32];
    float n2 = x2 * rstd * w[lane + 64], n3 = x3 * rstd * w[lane + 96];
    out[lane]      = f2bf(n0 * c0 - n2 * s0);
    out[lane + 32] = f2bf(n1 * c1 - n3 * s1);
    out[lane + 64] = f2bf(n2 * c0 + n0 * s0);
    out[lane + 96] = f2bf(n3 * c1 + n1 * s1);
}

__global__ __launch_bounds__(128) void norm_rope(const float* __restrict__ qkv,
                                                 const int* __restrict__ pos,
                                                 const float* __restrict__ qw,
                                                 const float* __restrict__ kw,
                                                 bf16_t* __restrict__ qo,
                                                 bf16_t* __restrict__ ko,
                                                 bf16_t* __restrict__ vo) {
    const int t = blockIdx.x, tid = threadIdx.x, lane = tid & 31, wave = tid >> 5;
    const float p = (float)pos[t];
    const float* base = qkv + (size_t)t * QKV_N;
    float inv0 = powf(10000.0f, -(float)lane * (1.0f / 64.0f));
    float inv1 = powf(10000.0f, -(float)(lane + 32) * (1.0f / 64.0f));
    float a0 = p * inv0, a1 = p * inv1;
    float c0 = cosf(a0), s0 = sinf(a0), c1 = cosf(a1), s1 = sinf(a1);

    for (int hh = 0; hh < 8; ++hh) {
        int head = wave * 8 + hh;
        rmsrope_store(base + head * HDIM, qw,
                      qo + (size_t)t * QSIZE + head * HDIM, lane, c0, s0, c1, s1);
    }
    rmsrope_store(base + QSIZE + wave * HDIM, kw,
                  ko + (size_t)t * KVSIZE + wave * HDIM, lane, c0, s0, c1, s1);
    for (int i = tid; i < KVSIZE; i += 128)
        vo[(size_t)t * KVSIZE + i] = f2bf(base[QSIZE + KVSIZE + i]);
}

// -------------------------------------------- causal flash attention (wave32)
// grid (T/64, NHEADS), 128 threads = 4 waves, each wave owns 16 q-rows.
// K tile staged by TDM (pad 128->136 halves); V transposed manually at stage.
__global__ __launch_bounds__(128) void flash_attn(const bf16_t* __restrict__ Q,
                                                  const bf16_t* __restrict__ Kc,
                                                  const bf16_t* __restrict__ Vc,
                                                  bf16_t* __restrict__ Oout) {
    __shared__ bf16_t Kt[2][32 * 136];   // [key][d], pad 136
    __shared__ bf16_t VT[2][128 * 40];   // [d][key], pad 40 (transposed at stage)
    __shared__ bf16_t Pb[4 * 16 * 32];   // per-wave P tile, row stride 32

    const int tid = threadIdx.x, lane = tid & 31, wave = tid >> 5;
    const int q0 = blockIdx.x * 64, h = blockIdx.y, kvh = h >> 3;
    const int aoff = (lane & 16) ? 8 : 0;
    const int bsel = (lane & 16) ? 16 : 0;
    const int mycol = lane & 15;
    const int rowadd = (lane & 16) ? 8 : 0;
    const float scale = 0.08838834764831845f;      // 1/sqrt(128)
    const float L2E = 1.4426950408889634f;

    // Q fragments (row per lane), resident for the whole pass
    const int qrow = q0 + wave * 16 + mycol;
    v16bf qf[4];
    for (int c = 0; c < 4; ++c) {
        const bf16_t* gp = Q + (size_t)qrow * QSIZE + h * HDIM + c * 32 + aoff;
        v8bf lo = *(const v8bf*)gp;
        v8bf hi = *(const v8bf*)(gp + 16);
        qf[c] = __builtin_shufflevector(lo, hi, 0,1,2,3,4,5,6,7,8,9,10,11,12,13,14,15);
    }

    union uv8 { v8f v; float f[8]; };
    uv8 Ov[8];
    v8f z = {0.f,0.f,0.f,0.f,0.f,0.f,0.f,0.f};
    float mrow[8], lrow[8];
    for (int c = 0; c < 8; ++c) Ov[c].v = z;
    for (int j = 0; j < 8; ++j) { mrow[j] = -3e38f; lrow[j] = 0.f; }

    auto stage = [&](int buf, int k0) {
#if USE_TDM
        if (tid < 32)
            tdm_load_2d((unsigned)(size_t)&Kt[buf][0],
                        Kc + (size_t)k0 * KVSIZE + kvh * HDIM,
                        128u, 32u, (unsigned)KVSIZE, 5u /*64 DW*/, 3u /*4 DW*/);
        for (int i = 0; i < 4; ++i) {   // V transposed by VALU (TDM cannot transpose)
            int idx = tid + i * 128;
            int row = idx >> 4, c = idx & 15;
            union { v8bf v; bf16_t s[8]; } vv;
            vv.v = *(const v8bf*)(Vc + (size_t)(k0 + row) * KVSIZE + kvh * HDIM + c * 8);
            for (int e = 0; e < 8; ++e)
                VT[buf][(c * 8 + e) * 40 + row] = vv.s[e];
        }
#else
        for (int i = 0; i < 4; ++i) {
            int idx = tid + i * 128;
            int row = idx >> 4, c = idx & 15;
            *(v8bf*)&Kt[buf][row * 136 + c * 8] =
                *(const v8bf*)(Kc + (size_t)(k0 + row) * KVSIZE + kvh * HDIM + c * 8);
            union { v8bf v; bf16_t s[8]; } vv;
            vv.v = *(const v8bf*)(Vc + (size_t)(k0 + row) * KVSIZE + kvh * HDIM + c * 8);
            for (int e = 0; e < 8; ++e)
                VT[buf][(c * 8 + e) * 40 + row] = vv.s[e];
        }
#endif
    };

    const int nkt = (q0 >> 5) + 2;   // cover keys 0 .. q0+63
    stage(0, 0);
    for (int kt = 0; kt < nkt; ++kt) {
        const int k0 = kt * 32;
#if USE_TDM
        __builtin_amdgcn_s_wait_tensorcnt(0);
#endif
        __syncthreads();
        if (kt + 1 < nkt) stage((kt + 1) & 1, (kt + 1) * 32);
        const bf16_t* Kb = Kt[kt & 1];
        const bf16_t* Vb = VT[kt & 1];

        // S = Q * K^T   (two 16x16 key tiles, contract over d in 4 steps)
        v8f S0 = z, S1 = z;
        for (int c = 0; c < 4; ++c) {
            v16bf b0 = frag_b(&Kb[mycol * 136 + c * 32], bsel);
            v16bf b1 = frag_b(&Kb[(16 + mycol) * 136 + c * 32], bsel);
            S0 = wmma_bf16(qf[c], b0, S0);
            S1 = wmma_bf16(qf[c], b1, S1);
        }
        uv8 u0, u1; u0.v = S0; u1.v = S1;

        bf16_t* myP = &Pb[wave * 512];
        for (int j = 0; j < 8; ++j) {
            int qidx = q0 + wave * 16 + j + rowadd;
            float s0v = u0.f[j] * scale; if (k0 + mycol > qidx)      s0v = -3e38f;
            float s1v = u1.f[j] * scale; if (k0 + 16 + mycol > qidx) s1v = -3e38f;
            float mx = fmaxf(s0v, s1v);
            for (int m = 1; m < 16; m <<= 1) mx = fmaxf(mx, __shfl_xor(mx, m, 32));
            float newm = fmaxf(mrow[j], mx);
            float alpha = exp2f((mrow[j] - newm) * L2E);
            mrow[j] = newm;
            float p0 = exp2f((s0v - newm) * L2E);
            float p1 = exp2f((s1v - newm) * L2E);
            float ps = p0 + p1;
            for (int m = 1; m < 16; m <<= 1) ps += __shfl_xor(ps, m, 32);
            lrow[j] = lrow[j] * alpha + ps;
            for (int c = 0; c < 8; ++c) Ov[c].f[j] *= alpha;
            int prow = j + rowadd;
            myP[prow * 32 + mycol]      = f2bf(p0);
            myP[prow * 32 + 16 + mycol] = f2bf(p1);
        }

        // O += P * V   (contract over 32 keys, 8 d-column tiles)
        v16bf pf = frag_a(&myP[mycol * 32], aoff);
        for (int c8 = 0; c8 < 8; ++c8) {
            v16bf bv = frag_b(&Vb[(c8 * 16 + mycol) * 40], bsel);
            Ov[c8].v = wmma_bf16(pf, bv, Ov[c8].v);
        }
    }

    for (int c8 = 0; c8 < 8; ++c8)
        for (int j = 0; j < 8; ++j) {
            int row = q0 + wave * 16 + j + rowadd;
            float val = Ov[c8].f[j] / lrow[j];
            Oout[(size_t)row * QSIZE + h * HDIM + c8 * 16 + mycol] = f2bf(val);
        }
}

// ---------------------------------------------------------------------- launch
extern "C" void kernel_launch(void* const* d_in, const int* in_sizes, int n_in,
                              void* d_out, int out_size, void* d_ws, size_t ws_size,
                              hipStream_t stream) {
    const int*   positions = (const int*)d_in[0];
    const float* hidden    = (const float*)d_in[1];
    const float* w_qkv     = (const float*)d_in[2];
    const float* w_o       = (const float*)d_in[3];
    const float* q_norm    = (const float*)d_in[4];
    const float* k_norm    = (const float*)d_in[5];
    float* out = (float*)d_out;

    char* ws = (char*)d_ws;
    size_t off = 0;
    bf16_t* hidden_bf = (bf16_t*)(ws + off); off += (size_t)T_TOK * HIDDEN * 2;
    bf16_t* wqkv_bf   = (bf16_t*)(ws + off); off += (size_t)QKV_N * HIDDEN * 2;
    bf16_t* wo_bf     = (bf16_t*)(ws + off); off += (size_t)HIDDEN * QSIZE * 2;
    float*  qkv_f32   = (float*) (ws + off); off += (size_t)T_TOK * QKV_N * 4;
    bf16_t* q_bf      = (bf16_t*)(ws + off); off += (size_t)T_TOK * QSIZE * 2;
    bf16_t* k_bf      = (bf16_t*)(ws + off); off += (size_t)T_TOK * KVSIZE * 2;
    bf16_t* v_bf      = (bf16_t*)(ws + off); off += (size_t)T_TOK * KVSIZE * 2;
    bf16_t* attn_bf   = (bf16_t*)(ws + off); off += (size_t)T_TOK * QSIZE * 2;

    cvt_f32_bf16<<<1024, 256, 0, stream>>>(hidden, hidden_bf, T_TOK * HIDDEN / 4);
    cvt_f32_bf16<<<1024, 256, 0, stream>>>(w_qkv, wqkv_bf, QKV_N * HIDDEN / 4);
    cvt_f32_bf16<<<1024, 256, 0, stream>>>(w_o, wo_bf, HIDDEN * QSIZE / 4);

    gemm_bf16_nt<<<dim3(QKV_N / 128, T_TOK / 128), 256, 0, stream>>>(
        hidden_bf, wqkv_bf, qkv_f32, T_TOK, QKV_N, HIDDEN);

    norm_rope<<<T_TOK, 128, 0, stream>>>(qkv_f32, positions, q_norm, k_norm,
                                         q_bf, k_bf, v_bf);

    flash_attn<<<dim3(T_TOK / 64, NHEADS), 128, 0, stream>>>(q_bf, k_bf, v_bf, attn_bf);

    gemm_bf16_nt<<<dim3(HIDDEN / 128, T_TOK / 128), 256, 0, stream>>>(
        attn_bf, wo_bf, out, T_TOK, HIDDEN, QSIZE);
}
